// QuantizedLinear_88673894793301
// MI455X (gfx1250) — compile-verified, matrix operand order checked by reading, not run on hardware
//
#include <hip/hip_runtime.h>

// ---------------------------------------------------------------------------
// Quantized (group-wise INT4) linear:  out = x @ dequant(W)^T + bias
//   x [4,2048,4096] f32 -> M=8192, K=4096 ; Wq [4096,4096] i32 (int4 vals)
//   group g = o*32 + (i>>7), size 128 along K ; out [4,2048,4096] f32
//
// bf16 WMMA, f32 accum.  Block tile 256(M) x 128(N) x 64(K), 512 threads
// = 16 wave32 waves in a 4x4 grid, wave tile 64x32 (4x2 WMMA accums).
// Double-buffered LDS (one barrier / K-tile), global->reg prefetch of the
// next tile overlapped with WMMA compute, on-the-fly int4 dequant to bf16.
// MAC/VALU ratio ~73 keeps the matrix pipes (XDL, co-executing with VALU)
// the bottleneck instead of the convert/dequant VALU work.
// ---------------------------------------------------------------------------

typedef __attribute__((ext_vector_type(16))) __bf16 bf16x16;
typedef __attribute__((ext_vector_type(8)))  __bf16 bf16x8;
typedef __attribute__((ext_vector_type(2)))  __bf16 bf16x2;
typedef __attribute__((ext_vector_type(8)))  float  v8f;
typedef __attribute__((ext_vector_type(4)))  float  f32x4;
typedef __attribute__((ext_vector_type(4)))  int    i32x4;

constexpr int Mdim = 8192;
constexpr int Ndim = 4096;
constexpr int Kdim = 4096;

constexpr int BM = 256;
constexpr int BN = 128;
constexpr int BK = 64;
constexpr int LDSA = BK + 8;        // ushort stride: 144B rows, conflict-free b128
constexpr int LDSB = BK + 8;
constexpr int KTILES = Kdim / BK;   // 64
constexpr int NTHREADS = 512;       // 16 wave32 waves

// hardware bf16 convert path: fptrunc f32->bf16 (gfx1250 has native bf16 cvt)
__device__ inline unsigned int pack2_bf16(float a, float b) {
  bf16x2 p;
  p.x = (__bf16)a;
  p.y = (__bf16)b;
  return __builtin_bit_cast(unsigned int, p);
}

__global__ __launch_bounds__(NTHREADS)
void qlinear_wmma_bf16_kernel(const float* __restrict__ X,
                              const int*   __restrict__ Wq,
                              const float* __restrict__ scales,
                              const int*   __restrict__ zps,
                              const float* __restrict__ bias,
                              float*       __restrict__ Out) {
  __shared__ unsigned short lds_a[2][BM * LDSA];  // 2 * 36 KB
  __shared__ unsigned short lds_b[2][BN * LDSB];  // 2 * 18 KB

  const int tid  = threadIdx.x;
  const int lane = tid & 31;
  const int wid  = tid >> 5;       // 0..15
  const int wm   = wid & 3;        // wave M slot (4) -> 64 rows each
  const int wn   = wid >> 2;       // wave N slot (4) -> 32 cols each
  const int bm   = blockIdx.y * BM;
  const int bn   = blockIdx.x * BN;

  const int lrow  = lane & 15;     // row within 16x16 fragment
  const int khalf = lane >> 4;     // K-half selector (bf16 WMMA A/B layout)

  v8f acc[4][2];
#pragma unroll
  for (int mi = 0; mi < 4; ++mi)
#pragma unroll
    for (int ni = 0; ni < 2; ++ni)
      acc[mi][ni] = v8f{};

  f32x4 areg[8];                   // 256x64 f32 / 512 thr = 8 float4
  i32x4 breg[4];                   // 128x64 i32 / 512 thr = 4 int4
  float bscale[4];
  float bzp[4];

  // ---- stage tile 0 ----
  {
    const int k0 = 0;
#pragma unroll
    for (int i = 0; i < 8; ++i) {
      const int idx = tid + i * NTHREADS;
      const int row = idx >> 4;          // 0..255
      const int c4  = idx & 15;
      areg[i] = *(const f32x4*)(X + (size_t)(bm + row) * Kdim + k0 + c4 * 4);
    }
#pragma unroll
    for (int i = 0; i < 4; ++i) {
      const int idx = tid + i * NTHREADS;
      const int row = idx >> 4;          // 0..127
      const int c4  = idx & 15;
      breg[i] = *(const i32x4*)(Wq + (size_t)(bn + row) * Kdim + k0 + c4 * 4);
      const int g = (bn + row) * (Kdim / 128) + (k0 >> 7);
      bscale[i] = scales[g];
      bzp[i]    = (float)zps[g];
    }
  }
  // ---- store tile 0 to buffer 0 ----
#pragma unroll
  for (int i = 0; i < 8; ++i) {
    const int idx = tid + i * NTHREADS;
    const int row = idx >> 4;
    const int c4  = idx & 15;
    unsigned int* dst = (unsigned int*)&lds_a[0][row * LDSA + c4 * 4];
    dst[0] = pack2_bf16(areg[i].x, areg[i].y);
    dst[1] = pack2_bf16(areg[i].z, areg[i].w);
  }
#pragma unroll
  for (int i = 0; i < 4; ++i) {
    const int idx = tid + i * NTHREADS;
    const int row = idx >> 4;
    const int c4  = idx & 15;
    const float s = bscale[i];
    const float z = bzp[i];
    unsigned int* dst = (unsigned int*)&lds_b[0][row * LDSB + c4 * 4];
    dst[0] = pack2_bf16(((float)breg[i].x - z) * s, ((float)breg[i].y - z) * s);
    dst[1] = pack2_bf16(((float)breg[i].z - z) * s, ((float)breg[i].w - z) * s);
  }
  __syncthreads();

  for (int kt = 0; kt < KTILES; ++kt) {
    const int p = kt & 1;
    const bool more = (kt + 1) < KTILES;

    // ---- stage next tile into registers (overlaps WMMA compute below) ----
    if (more) {
      const int k0 = (kt + 1) * BK;
#pragma unroll
      for (int i = 0; i < 8; ++i) {
        const int idx = tid + i * NTHREADS;
        const int row = idx >> 4;
        const int c4  = idx & 15;
        areg[i] = *(const f32x4*)(X + (size_t)(bm + row) * Kdim + k0 + c4 * 4);
      }
#pragma unroll
      for (int i = 0; i < 4; ++i) {
        const int idx = tid + i * NTHREADS;
        const int row = idx >> 4;
        const int c4  = idx & 15;
        breg[i] = *(const i32x4*)(Wq + (size_t)(bn + row) * Kdim + k0 + c4 * 4);
        const int g = (bn + row) * (Kdim / 128) + (k0 >> 7);
        bscale[i] = scales[g];
        bzp[i]    = (float)zps[g];
      }
      // pull kt+2 toward L2 (global_prefetch_b8)
      if (kt + 2 < KTILES) {
        __builtin_prefetch(X  + (size_t)(bm + (tid >> 1)) * Kdim + k0 + BK + (tid & 1) * 32, 0, 2);
        __builtin_prefetch(Wq + (size_t)(bn + (tid & 127)) * Kdim + k0 + BK + (tid >> 7) * 16, 0, 2);
      }
    }

    // ---- compute from buffer p: two k=32 WMMA steps ----
#pragma unroll
    for (int ks = 0; ks < BK; ks += 32) {
      bf16x16 afrag[4];
      bf16x16 bfrag[2];
#pragma unroll
      for (int mi = 0; mi < 4; ++mi) {
        const int row = wm * 64 + mi * 16 + lrow;
        const bf16x8 lo = *(const bf16x8*)&lds_a[p][row * LDSA + ks + khalf * 8];
        const bf16x8 hi = *(const bf16x8*)&lds_a[p][row * LDSA + ks + khalf * 8 + 16];
        afrag[mi] = __builtin_shufflevector(lo, hi,
            0,1,2,3,4,5,6,7,8,9,10,11,12,13,14,15);
      }
#pragma unroll
      for (int ni = 0; ni < 2; ++ni) {
        const int row = wn * 32 + ni * 16 + lrow;   // row = output feature
        const bf16x8 lo = *(const bf16x8*)&lds_b[p][row * LDSB + ks + khalf * 8];
        const bf16x8 hi = *(const bf16x8*)&lds_b[p][row * LDSB + ks + khalf * 8 + 16];
        bfrag[ni] = __builtin_shufflevector(lo, hi,
            0,1,2,3,4,5,6,7,8,9,10,11,12,13,14,15);
      }
#pragma unroll
      for (int mi = 0; mi < 4; ++mi) {
#pragma unroll
        for (int ni = 0; ni < 2; ++ni) {
          acc[mi][ni] = __builtin_amdgcn_wmma_f32_16x16x32_bf16(
              false, afrag[mi], false, bfrag[ni],
              (short)0, acc[mi][ni], false, false);
        }
      }
    }

    // ---- drain staged registers into the other buffer ----
    if (more) {
      const int q = p ^ 1;
#pragma unroll
      for (int i = 0; i < 8; ++i) {
        const int idx = tid + i * NTHREADS;
        const int row = idx >> 4;
        const int c4  = idx & 15;
        unsigned int* dst = (unsigned int*)&lds_a[q][row * LDSA + c4 * 4];
        dst[0] = pack2_bf16(areg[i].x, areg[i].y);
        dst[1] = pack2_bf16(areg[i].z, areg[i].w);
      }
#pragma unroll
      for (int i = 0; i < 4; ++i) {
        const int idx = tid + i * NTHREADS;
        const int row = idx >> 4;
        const int c4  = idx & 15;
        const float s = bscale[i];
        const float z = bzp[i];
        unsigned int* dst = (unsigned int*)&lds_b[q][row * LDSB + c4 * 4];
        dst[0] = pack2_bf16(((float)breg[i].x - z) * s, ((float)breg[i].y - z) * s);
        dst[1] = pack2_bf16(((float)breg[i].z - z) * s, ((float)breg[i].w - z) * s);
      }
    }
    __syncthreads();
  }

  // ---- epilogue: C layout (VGPR r: M=r / r+8 by lane half; N=lane&15) ----
#pragma unroll
  for (int mi = 0; mi < 4; ++mi) {
#pragma unroll
    for (int ni = 0; ni < 2; ++ni) {
      const int n  = bn + wn * 32 + ni * 16 + (lane & 15);
      const float bv = bias[n];
      const int mb = bm + wm * 64 + mi * 16 + khalf * 8;
#pragma unroll
      for (int r = 0; r < 8; ++r) {
        Out[(size_t)(mb + r) * Ndim + n] = acc[mi][ni][r] + bv;
      }
    }
  }
}

extern "C" void kernel_launch(void* const* d_in, const int* in_sizes, int n_in,
                              void* d_out, int out_size, void* d_ws, size_t ws_size,
                              hipStream_t stream) {
  const float* x      = (const float*)d_in[0];   // [4,2048,4096] f32
  const int*   wq     = (const int*)  d_in[1];   // [4096,4096] int32 (int4 values)
  const float* scales = (const float*)d_in[2];   // [131072] f32
  const int*   zps    = (const int*)  d_in[3];   // [131072] int32
  const float* bias   = (const float*)d_in[4];   // [4096] f32
  float* out = (float*)d_out;                    // [4,2048,4096] f32

  (void)in_sizes; (void)n_in; (void)out_size; (void)d_ws; (void)ws_size;

  dim3 grid(Ndim / BN, Mdim / BM);   // (32, 32)
  dim3 block(NTHREADS);              // 16 wave32 waves
  qlinear_wmma_bf16_kernel<<<grid, block, 0, stream>>>(x, wq, scales, zps, bias, out);
}
